// Conv_LSTM_70729521430836
// MI455X (gfx1250) — compile-verified, matrix operand order checked by reading, not run on hardware
//
#include <hip/hip_runtime.h>
#include <hip/hip_bf16.h>

typedef __attribute__((ext_vector_type(16))) __bf16 v16bf;
typedef __attribute__((ext_vector_type(8)))  __bf16 v8bf;
typedef __attribute__((ext_vector_type(8)))  float  v8f;

#define BB   16
#define TT   270
#define NT   (BB*TT)      // 4320 frames
#define HIDN 256
#define G4   1024         // 4*HID
#define KF   6272         // 32*14*14

// ---------------------------------------------------------------------------
// helpers
// ---------------------------------------------------------------------------
__device__ __forceinline__ float sigm(float x) { return 1.0f / (1.0f + __expf(-x)); }

__device__ __forceinline__ v8bf zero8() {
    v8bf z;
    #pragma unroll
    for (int i = 0; i < 8; ++i) z[i] = (__bf16)0.0f;
    return z;
}

__device__ __forceinline__ v16bf combine(v8bf lo, v8bf hi) {
    return __builtin_shufflevector(lo, hi, 0,1,2,3,4,5,6,7,8,9,10,11,12,13,14,15);
}

// 16x32 bf16 fragment: lane holds 8 contiguous K at (half*8) and 8 at (half*8+16)
__device__ __forceinline__ v16bf load_frag(const __bf16* p) {
    return combine(*(const v8bf*)(p), *(const v8bf*)(p + 16));
}

__device__ __forceinline__ v8f wmma_bf16(v16bf a, v16bf b, v8f c) {
    return __builtin_amdgcn_wmma_f32_16x16x32_bf16(false, a, false, b, (short)0, c, false, false);
}

// async global -> LDS copy (16B per lane), tracked by ASYNCcnt
__device__ __forceinline__ void async_b128(unsigned lds_byte_off, const void* gptr) {
    asm volatile("global_load_async_to_lds_b128 %0, %1, off"
                 :: "v"(lds_byte_off), "v"(gptr) : "memory");
}

// ---------------------------------------------------------------------------
// fp32 -> bf16 conversion
// ---------------------------------------------------------------------------
__global__ void f2bf_kernel(const float* __restrict__ in, __bf16* __restrict__ out, int n) {
    int i = blockIdx.x * 256 + threadIdx.x;
    if (i < n) out[i] = (__bf16)in[i];
}

// reorder w_conv2 [32][16][3][3] -> [co][ (ky*3+kx)*16 + ci ] bf16 (im2col K order)
__global__ void w2_reorg_kernel(const float* __restrict__ w2, __bf16* __restrict__ w2c) {
    int i = blockIdx.x * 256 + threadIdx.x;
    if (i < 32 * 144) {
        int co = i / 144, k = i % 144;
        int e = k >> 4, ci = k & 15;
        int ky = e / 3, kx = e - ky * 3;
        w2c[i] = (__bf16)w2[((co * 16 + ci) * 3 + ky) * 3 + kx];
    }
}

// ---------------------------------------------------------------------------
// conv1 (3->16, 3x3 VALID) + relu + 2x2 maxpool
// [NT,3,64,64] fp32 -> o1c[n][y][x][ci] bf16 (channels-last, 31x31x16)
// ---------------------------------------------------------------------------
__global__ void __launch_bounds__(256) conv1_pool_kernel(
    const float* __restrict__ x, const float* __restrict__ w,
    const float* __restrict__ bias, __bf16* __restrict__ o1c)
{
    int n  = blockIdx.x >> 4;
    int co = blockIdx.x & 15;
    __shared__ float wsm[27];
    if (threadIdx.x < 27) wsm[threadIdx.x] = w[co * 27 + threadIdx.x];
    __syncthreads();
    float bv = bias[co];
    const float* xin = x + (size_t)n * 3 * 64 * 64;

    for (int idx = threadIdx.x; idx < 961; idx += 256) {
        int py = idx / 31, px = idx % 31;
        float m = -3.4e38f;
        #pragma unroll
        for (int dy = 0; dy < 2; ++dy) {
            #pragma unroll
            for (int dx = 0; dx < 2; ++dx) {
                int oy = 2 * py + dy, ox = 2 * px + dx;
                float s = bv;
                #pragma unroll
                for (int ci = 0; ci < 3; ++ci)
                    #pragma unroll
                    for (int ky = 0; ky < 3; ++ky)
                        #pragma unroll
                        for (int kx = 0; kx < 3; ++kx)
                            s += xin[ci * 4096 + (oy + ky) * 64 + (ox + kx)] * wsm[ci * 9 + ky * 3 + kx];
                s = fmaxf(s, 0.0f);
                m = fmaxf(m, s);
            }
        }
        o1c[(size_t)((n * 31 + py) * 31 + px) * 16 + co] = (__bf16)m;
    }
}

// ---------------------------------------------------------------------------
// conv2 as im2col WMMA GEMM: M = 784 positions/frame (28x28), N = 32, K = 144
// (K padded to 160; last hi-chunk zero-filled uniformly).  Raw sums -> c2 bf16.
// block = 7 waves, wave = one 16-row tile x 32 cols; 49 tiles per frame.
// ---------------------------------------------------------------------------
__device__ __forceinline__ v8bf ld_chunkA(const __bf16* abase, int kc) {
    if (kc >= 144) return zero8();
    int e = kc >> 4, ci0 = kc & 15;
    int ky = e / 3, kx = e - ky * 3;
    return *(const v8bf*)(abase + (ky * 31 + kx) * 16 + ci0);
}

__global__ void __launch_bounds__(224) conv2_wmma_kernel(
    const __bf16* __restrict__ o1c, const __bf16* __restrict__ w2c,
    __bf16* __restrict__ c2)   // c2[n][pos(784)][co(32)]
{
    int n = blockIdx.x / 7, g = blockIdx.x % 7;
    int wave = threadIdx.x >> 5, lane = threadIdx.x & 31;
    int hf = lane >> 4, nl = lane & 15;
    int tile = g * 7 + wave;            // 0..48
    int rr = tile * 16 + nl;            // A row for this lane (0..783)
    int oy = rr / 28, ox = rr % 28;
    const __bf16* abase = o1c + (size_t)((n * 31 + oy) * 31 + ox) * 16;

    v8f acc[2] = {v8f{}, v8f{}};
    #pragma unroll
    for (int k0 = 0; k0 < 160; k0 += 32) {
        int kcl = k0 + hf * 8, kch = kcl + 16;
        v16bf a = combine(ld_chunkA(abase, kcl), ld_chunkA(abase, kch));
        #pragma unroll
        for (int j = 0; j < 2; ++j) {
            const __bf16* brow = w2c + (j * 16 + nl) * 144;
            v8bf blo = (kcl < 144) ? *(const v8bf*)(brow + kcl) : zero8();
            v8bf bhi = (kch < 144) ? *(const v8bf*)(brow + kch) : zero8();
            acc[j] = wmma_bf16(a, combine(blo, bhi), acc[j]);
        }
    }
    #pragma unroll
    for (int j = 0; j < 2; ++j)
        #pragma unroll
        for (int r = 0; r < 8; ++r)
            c2[((size_t)n * 784 + tile * 16 + hf * 8 + r) * 32 + j * 16 + nl] = (__bf16)acc[j][r];
}

// bias + relu + 2x2 maxpool -> featsB[n][co*196 + py*14 + px] bf16
__global__ void __launch_bounds__(256) pool2_kernel(
    const __bf16* __restrict__ c2, const float* __restrict__ b2,
    __bf16* __restrict__ featsB)
{
    long long idx = (long long)blockIdx.x * 256 + threadIdx.x;
    if (idx >= (long long)NT * KF) return;
    int n = (int)(idx / KF), f = (int)(idx % KF);
    int co = f / 196, pp = f % 196;
    int py = pp / 14, px = pp % 14;
    const __bf16* base = c2 + ((size_t)n * 784 + (2 * py) * 28 + 2 * px) * 32 + co;
    float bv = b2[co];
    float m = -3.4e38f;
    #pragma unroll
    for (int dy = 0; dy < 2; ++dy)
        #pragma unroll
        for (int dx = 0; dx < 2; ++dx) {
            float v = (float)base[(dy * 28 + dx) * 32] + bv;
            m = fmaxf(m, fmaxf(v, 0.0f));
        }
    featsB[idx] = (__bf16)m;
}

// ---------------------------------------------------------------------------
// gx = feats @ w_ih.T + b_ih  (M=4320, N=1024, K=6272)
// block = 8 waves sharing a 64-col B slab double-buffered in LDS via
// global_load_async_to_lds_b128 (ASYNCcnt); wave = 16x64 tile, bf16 WMMA.
// grid = (16 n-groups, 34 m-groups); tail waves compute but skip store.
// ---------------------------------------------------------------------------
__global__ void __launch_bounds__(256) gemm_gx_kernel(
    const __bf16* __restrict__ A,   // [4320][6272]
    const __bf16* __restrict__ Bm,  // [1024][6272] (row n = output col, K-contig)
    const float* __restrict__ bih, float* __restrict__ gx)
{
    __shared__ __bf16 Bs[2][64 * 32];          // 2 x 4KB
    int tid = threadIdx.x;
    int wave = tid >> 5, lane = tid & 31;
    int hf = lane >> 4, nl = lane & 15;
    int n0 = blockIdx.x * 64;
    int mtile = blockIdx.y * 8 + wave;
    bool valid = (mtile < 270);
    int m0 = (valid ? mtile : 269) * 16;

    // async producer: each thread copies one 16B chunk per buffer fill
    int row = tid >> 2, cc = tid & 3;          // row 0..63, chunk 0..3
    const __bf16* gB = Bm + (size_t)(n0 + row) * KF + cc * 8;
    unsigned ldsOff0 = (unsigned)(uintptr_t)(&Bs[0][row * 32 + cc * 8]);
    unsigned ldsOff1 = (unsigned)(uintptr_t)(&Bs[1][row * 32 + cc * 8]);

    const __bf16* pa = A + (size_t)(m0 + nl) * KF + hf * 8;

    v8f acc[4] = {v8f{}, v8f{}, v8f{}, v8f{}};

    async_b128(ldsOff0, gB);                   // prologue: buf0 <- k0=0
    int buf = 0;
    for (int k0 = 0; k0 < KF - 32; k0 += 32) {
        async_b128(buf ? ldsOff0 : ldsOff1, gB + k0 + 32);
        asm volatile("s_wait_asynccnt 0x1" ::: "memory");   // older fill done
        __syncthreads();
        __builtin_prefetch(pa + k0 + 64, 0, 0);
        v16bf a = load_frag(pa + k0);
        const __bf16* bs = &Bs[buf][0];
        v16bf b0 = load_frag(bs + (0 * 16 + nl) * 32 + hf * 8);
        v16bf b1 = load_frag(bs + (1 * 16 + nl) * 32 + hf * 8);
        v16bf b2 = load_frag(bs + (2 * 16 + nl) * 32 + hf * 8);
        v16bf b3 = load_frag(bs + (3 * 16 + nl) * 32 + hf * 8);
        acc[0] = wmma_bf16(a, b0, acc[0]);
        acc[1] = wmma_bf16(a, b1, acc[1]);
        acc[2] = wmma_bf16(a, b2, acc[2]);
        acc[3] = wmma_bf16(a, b3, acc[3]);
        __syncthreads();
        buf ^= 1;
    }
    // epilogue: last k-step (k0 = KF-32) in Bs[buf]
    asm volatile("s_wait_asynccnt 0x0" ::: "memory");
    __syncthreads();
    {
        v16bf a = load_frag(pa + (KF - 32));
        const __bf16* bs = &Bs[buf][0];
        v16bf b0 = load_frag(bs + (0 * 16 + nl) * 32 + hf * 8);
        v16bf b1 = load_frag(bs + (1 * 16 + nl) * 32 + hf * 8);
        v16bf b2 = load_frag(bs + (2 * 16 + nl) * 32 + hf * 8);
        v16bf b3 = load_frag(bs + (3 * 16 + nl) * 32 + hf * 8);
        acc[0] = wmma_bf16(a, b0, acc[0]);
        acc[1] = wmma_bf16(a, b1, acc[1]);
        acc[2] = wmma_bf16(a, b2, acc[2]);
        acc[3] = wmma_bf16(a, b3, acc[3]);
    }
    if (valid) {
        #pragma unroll
        for (int j = 0; j < 4; ++j) {
            int col = n0 + j * 16 + nl;
            float bv = bih[col];
            #pragma unroll
            for (int r = 0; r < 8; ++r)
                gx[(size_t)(m0 + hf * 8 + r) * G4 + col] = acc[j][r] + bv;
        }
    }
}

// ---------------------------------------------------------------------------
// LSTM scan: one persistent workgroup (32 wave32s).  h bf16 in LDS, gates fp32
// in LDS, cell state in registers; WMMA hidden projection each step.
// ---------------------------------------------------------------------------
__global__ void __launch_bounds__(1024, 1) lstm_scan_kernel(
    const float* __restrict__ gx,   // [4320][1024], includes b_ih
    const __bf16* __restrict__ Whh, // [1024][256]
    const float* __restrict__ bhh,
    const float* __restrict__ h0, const float* __restrict__ c0,
    float* __restrict__ hs)         // [4320][256]
{
    __shared__ float  g_s[16 * 1024];   // 64 KB
    __shared__ __bf16 h_s[16 * 256];    // 8 KB
    int tid  = threadIdx.x;
    int wave = tid >> 5, lane = tid & 31;
    int hf = lane >> 4, nl = lane & 15;
    int uu = tid & 255;
    int bq = tid >> 8;

    float c_reg[4];
    #pragma unroll
    for (int q = 0; q < 4; ++q) {
        int b = q * 4 + bq;
        c_reg[q] = c0[b * HIDN + uu];
        h_s[b * HIDN + uu] = (__bf16)h0[b * HIDN + uu];
    }
    __syncthreads();

    const __bf16* pb0 = Whh + (wave * 32 + nl) * HIDN + hf * 8;
    const __bf16* pa  = &h_s[nl * HIDN + hf * 8];

    for (int t = 0; t < TT; ++t) {
        v8f acc[2] = {v8f{}, v8f{}};
        #pragma unroll
        for (int k0 = 0; k0 < HIDN; k0 += 32) {
            v16bf a  = load_frag(pa + k0);
            v16bf b0 = load_frag(pb0 + k0);
            v16bf b1 = load_frag(pb0 + 16 * HIDN + k0);
            acc[0] = wmma_bf16(a, b0, acc[0]);
            acc[1] = wmma_bf16(a, b1, acc[1]);
        }
        int n0 = wave * 32;
        #pragma unroll
        for (int j = 0; j < 2; ++j)
            #pragma unroll
            for (int r = 0; r < 8; ++r)
                g_s[(hf * 8 + r) * G4 + n0 + j * 16 + nl] = acc[j][r];
        __syncthreads();

        #pragma unroll
        for (int q = 0; q < 4; ++q) {
            int b = q * 4 + bq;
            const float* gxr = gx + (size_t)(b * TT + t) * G4;
            float iv = g_s[b * G4 + uu]       + gxr[uu]       + bhh[uu];
            float fv = g_s[b * G4 + 256 + uu] + gxr[256 + uu] + bhh[256 + uu];
            float gv = g_s[b * G4 + 512 + uu] + gxr[512 + uu] + bhh[512 + uu];
            float ov = g_s[b * G4 + 768 + uu] + gxr[768 + uu] + bhh[768 + uu];
            float cn = sigm(fv) * c_reg[q] + sigm(iv) * tanhf(gv);
            float hn = sigm(ov) * tanhf(cn);
            c_reg[q] = cn;
            hs[(size_t)(b * TT + t) * HIDN + uu] = hn;
            h_s[b * HIDN + uu] = (__bf16)hn;
        }
        __syncthreads();
    }
}

// ---------------------------------------------------------------------------
// head: out[m] = dot(hs[m], w_head) + b_head ; one wave per output
// ---------------------------------------------------------------------------
__global__ void __launch_bounds__(256) head_kernel(
    const float* __restrict__ hs, const float* __restrict__ wh,
    const float* __restrict__ bh, float* __restrict__ out)
{
    int wave = threadIdx.x >> 5, lane = threadIdx.x & 31;
    int o = blockIdx.x * 8 + wave;
    const float* r = hs + (size_t)o * HIDN;
    float s = 0.0f;
    #pragma unroll
    for (int k = 0; k < 8; ++k) s += r[lane + k * 32] * wh[lane + k * 32];
    #pragma unroll
    for (int off = 16; off; off >>= 1) s += __shfl_xor(s, off, 32);
    if (lane == 0) out[o] = s + bh[0];
}

// ---------------------------------------------------------------------------
extern "C" void kernel_launch(void* const* d_in, const int* in_sizes, int n_in,
                              void* d_out, int out_size, void* d_ws, size_t ws_size,
                              hipStream_t stream) {
    const float* x     = (const float*)d_in[0];
    const float* w1    = (const float*)d_in[1];
    const float* b1    = (const float*)d_in[2];
    const float* w2    = (const float*)d_in[3];
    const float* b2    = (const float*)d_in[4];
    const float* wih   = (const float*)d_in[5];
    const float* whh   = (const float*)d_in[6];
    const float* bih   = (const float*)d_in[7];
    const float* bhh   = (const float*)d_in[8];
    const float* whead = (const float*)d_in[9];
    const float* bhead = (const float*)d_in[10];
    const float* h0    = (const float*)d_in[11];
    const float* c0    = (const float*)d_in[12];

    char* p = (char*)d_ws;
    __bf16* o1c    = (__bf16*)p;  p += (size_t)NT * 961 * 16 * 2;      // 132.8 MB
    __bf16* c2     = (__bf16*)p;  p += (size_t)NT * 784 * 32 * 2;      // 216.8 MB
    __bf16* featsB = (__bf16*)p;  p += (size_t)NT * KF * 2;            // 54.2 MB
    __bf16* wihB   = (__bf16*)p;  p += (size_t)G4 * KF * 2;            // 12.8 MB
    __bf16* whhB   = (__bf16*)p;  p += (size_t)G4 * HIDN * 2;          // 0.5 MB
    __bf16* w2c    = (__bf16*)p;  p += (size_t)32 * 144 * 2;           // 9 KB
    float*  gx     = (float*)p;   p += (size_t)NT * G4 * 4;            // 17.7 MB
    float*  hs     = (float*)p;   p += (size_t)NT * HIDN * 4;          // 4.4 MB

    f2bf_kernel<<<(G4 * KF + 255) / 256, 256, 0, stream>>>(wih, wihB, G4 * KF);
    f2bf_kernel<<<(G4 * HIDN + 255) / 256, 256, 0, stream>>>(whh, whhB, G4 * HIDN);
    w2_reorg_kernel<<<(32 * 144 + 255) / 256, 256, 0, stream>>>(w2, w2c);
    conv1_pool_kernel<<<NT * 16, 256, 0, stream>>>(x, w1, b1, o1c);
    conv2_wmma_kernel<<<NT * 7, 224, 0, stream>>>(o1c, w2c, c2);
    {
        long long tot = (long long)NT * KF;
        pool2_kernel<<<(unsigned)((tot + 255) / 256), 256, 0, stream>>>(c2, b2, featsB);
    }
    {
        dim3 grid(16, 34);
        gemm_gx_kernel<<<grid, 256, 0, stream>>>(featsB, wihB, bih, gx);
    }
    lstm_scan_kernel<<<1, 1024, 0, stream>>>(gx, whhB, bhh, h0, c0, hs);
    head_kernel<<<540, 256, 0, stream>>>(hs, whead, bhead, (float*)d_out);
}
